// Complex_Fully_Connected_Linear_Discriminator_LPF_32865089749150
// MI455X (gfx1250) — matrix operand order repaired and checked by reading, not compiled, not run on hardware
//
#include <hip/hip_runtime.h>
#include <cstddef>
#include <cstdint>

// ---------------------------------------------------------------------------
// MI455X (gfx1250) implementation: bf16 WMMA (f32 accumulate) fused complex
// GEMMs. All weights converted once to bf16 (fits in 192MB L2), activations
// kept bf16 between layers. v_wmma_f32_16x16x32_bf16 throughout.
// ---------------------------------------------------------------------------

typedef __bf16 bf16;
typedef __attribute__((ext_vector_type(16))) __bf16 v16bf;
typedef __attribute__((ext_vector_type(8)))  __bf16 v8bf;
typedef __attribute__((ext_vector_type(8)))  float  v8f;

#define HDIM   768
#define W2DIM  3072
#define BDIM   256
#define TDIM   64
#define MTOT   (BDIM * TDIM)   // 16384 rows for the pre-projection

__device__ __forceinline__ float lrelu(float v) { return v >= 0.0f ? v : 0.1f * v; }

// --- fragment loaders (wave32 WMMA 16-bit layouts, CDNA5 ISA) ----
// A (16x32, MxK): lanes 0-15 => M=lane, K in {kb..kb+7, 16+kb..16+kb+7}, kb=0
//                 lanes 16-31 => M=lane-16, kb=8. Two contiguous 16B loads.
__device__ __forceinline__ v16bf load_frag_a(const bf16* __restrict__ A, int ld,
                                             int row, int kc, int hi16) {
  const bf16* p = A + (size_t)row * ld + kc + hi16 * 8;
  v8bf lo = *(const v8bf*)p;          // K = kc+kb .. kc+kb+7
  v8bf hi = *(const v8bf*)(p + 16);   // K = kc+16+kb .. kc+16+kb+7
  return __builtin_shufflevector(lo, hi, 0,1,2,3,4,5,6,7,8,9,10,11,12,13,14,15);
}
// B (32x16, KxN) from row-major W[N][K] (computing X*W^T):
// lanes 0-15 => N=lane, K in {kc..kc+7, kc+16..kc+23} ; lanes 16-31 => N=lane-16, K in {kc+8..kc+15, kc+24..kc+31}
__device__ __forceinline__ v16bf load_frag_b(const bf16* __restrict__ W, int ld,
                                             int rowN, int kc, int hi16) {
  const bf16* p = W + (size_t)rowN * ld + kc + hi16 * 8;
  v8bf lo = *(const v8bf*)p;
  v8bf hi = *(const v8bf*)(p + 16);
  return __builtin_shufflevector(lo, hi, 0,1,2,3,4,5,6,7,8,9,10,11,12,13,14,15);
}

// ---------------------------------------------------------------------------
// Fused complex GEMM: computes rr=Ar*Wr^T, ii=Ai*Wi^T, ir=Ai*Wr^T, ri=Ar*Wi^T
// MODE 0: Or = rr-ii+(br-bi)            Oi = ir+ri+(br+bi)          (c-precompute)
// MODE 1: Or = lrelu(rr-ii+(br-bi)+Cr)  Oi = lrelu(ir+ri+(br+bi)+Ci) (RNN step)
// MODE 2: Or = lrelu(rr+br)-lrelu(ii+bi) Oi = lrelu(ir+br)+lrelu(ri+bi) (MLP)
// Block: 256 threads = 8 waves (2 M-waves x 4 N-waves); wave tile 32x16.
// Block tile 64(M) x 64(N). M,N multiples of 64; K multiple of 32.
// ---------------------------------------------------------------------------
template <int MODE>
__global__ __launch_bounds__(256)
void cgemm_kernel(const bf16* __restrict__ Ar, const bf16* __restrict__ Ai,
                  const bf16* __restrict__ Wr, const bf16* __restrict__ Wi,
                  const float* __restrict__ br, const float* __restrict__ bi,
                  const bf16* __restrict__ Cr, const bf16* __restrict__ Ci, int ldc,
                  bf16* __restrict__ Or, bf16* __restrict__ Oi,
                  int M, int N, int K) {
  const int tid  = threadIdx.x;
  const int wave = tid >> 5;
  const int lane = tid & 31;
  const int lr   = lane & 15;
  const int hi16 = lane >> 4;
  const int wm   = wave & 1;
  const int wn   = wave >> 1;
  const int m0   = blockIdx.y * 64 + wm * 32;
  const int n0   = blockIdx.x * 64 + wn * 16;

  v8f zero = {0.f, 0.f, 0.f, 0.f, 0.f, 0.f, 0.f, 0.f};
  v8f acc[4][2];   // [rr, ii, ir, ri][mi]
#pragma unroll
  for (int p = 0; p < 4; ++p)
#pragma unroll
    for (int mi = 0; mi < 2; ++mi) acc[p][mi] = zero;

  for (int kc = 0; kc < K; kc += 32) {
    if (kc + 32 < K) {  // prefetch streamed activations (global_prefetch_b8)
      __builtin_prefetch(Ar + (size_t)(m0 + lr) * K + kc + 32, 0, 1);
      __builtin_prefetch(Ai + (size_t)(m0 + lr) * K + kc + 32, 0, 1);
    }
    v16bf ar[2], ai[2];
#pragma unroll
    for (int mi = 0; mi < 2; ++mi) {
      ar[mi] = load_frag_a(Ar, K, m0 + mi * 16 + lr, kc, hi16);
      ai[mi] = load_frag_a(Ai, K, m0 + mi * 16 + lr, kc, hi16);
    }
    v16bf wrF = load_frag_b(Wr, K, n0 + lr, kc, hi16);
    v16bf wiF = load_frag_b(Wi, K, n0 + lr, kc, hi16);
#pragma unroll
    for (int mi = 0; mi < 2; ++mi) {
      acc[0][mi] = __builtin_amdgcn_wmma_f32_16x16x32_bf16(false, ar[mi], false, wrF, (short)0, acc[0][mi], false, false);
      acc[1][mi] = __builtin_amdgcn_wmma_f32_16x16x32_bf16(false, ai[mi], false, wiF, (short)0, acc[1][mi], false, false);
      acc[2][mi] = __builtin_amdgcn_wmma_f32_16x16x32_bf16(false, ai[mi], false, wrF, (short)0, acc[2][mi], false, false);
      acc[3][mi] = __builtin_amdgcn_wmma_f32_16x16x32_bf16(false, ar[mi], false, wiF, (short)0, acc[3][mi], false, false);
    }
  }

  // Epilogue. C/D layout: VGPR e, lanes 0-15 => M=e, lanes 16-31 => M=e+8; N=lane&15.
  const int   col = n0 + lr;
  const float vbr = br[col];
  const float vbi = bi[col];
#pragma unroll
  for (int mi = 0; mi < 2; ++mi) {
#pragma unroll
    for (int e = 0; e < 8; ++e) {
      const int row = m0 + mi * 16 + hi16 * 8 + e;
      const float rr = acc[0][mi][e];
      const float ii = acc[1][mi][e];
      const float ir = acc[2][mi][e];
      const float ri = acc[3][mi][e];
      float or_, oi_;
      if constexpr (MODE == 2) {
        or_ = lrelu(rr + vbr) - lrelu(ii + vbi);
        oi_ = lrelu(ir + vbr) + lrelu(ri + vbi);
      } else {
        or_ = rr - ii + (vbr - vbi);
        oi_ = ir + ri + (vbr + vbi);
        if constexpr (MODE == 1) {
          or_ += (float)Cr[(size_t)row * ldc + col];
          oi_ += (float)Ci[(size_t)row * ldc + col];
          or_ = lrelu(or_);
          oi_ = lrelu(oi_);
        }
      }
      Or[(size_t)row * N + col] = (bf16)or_;
      Oi[(size_t)row * N + col] = (bf16)oi_;
    }
  }
}

// --- f32 -> bf16 conversion (weights, h0) ----------------------------------
__global__ void cvt_kernel(const float* __restrict__ src, bf16* __restrict__ dst, int n) {
  int i = blockIdx.x * blockDim.x + threadIdx.x;
  if (i < n) dst[i] = (bf16)src[i];
}

// --- split x (B*T,1536) f32 into xr/xi (B*T,768) bf16 ----------------------
__global__ void split_x_kernel(const float* __restrict__ x,
                               bf16* __restrict__ xr, bf16* __restrict__ xi) {
  int i = blockIdx.x * blockDim.x + threadIdx.x;
  if (i >= MTOT * HDIM) return;
  int m = i / HDIM;
  int k = i - m * HDIM;
  const float* row = x + (size_t)m * (2 * HDIM);
  xr[i] = (bf16)row[k];
  xi[i] = (bf16)row[HDIM + k];
}

// --- final layer: out[m] = lrelu(sum xr*w[0:3072] + xi*w[3072:6144] + b) ---
__global__ void l5_kernel(const bf16* __restrict__ xr, const bf16* __restrict__ xi,
                          const float* __restrict__ w, const float* __restrict__ b,
                          float* __restrict__ out) {
  const int m = blockIdx.x;
  const int tid = threadIdx.x;
  float s = 0.0f;
  for (int j = tid; j < W2DIM; j += 256) {
    s += (float)xr[(size_t)m * W2DIM + j] * w[j];
    s += (float)xi[(size_t)m * W2DIM + j] * w[W2DIM + j];
  }
  __shared__ float red[256];
  red[tid] = s;
  __syncthreads();
  for (int off = 128; off > 0; off >>= 1) {
    if (tid < off) red[tid] += red[tid + off];
    __syncthreads();
  }
  if (tid == 0) {
    float v = red[0] + b[0];
    out[m] = lrelu(v);
  }
}

// ---------------------------------------------------------------------------
extern "C" void kernel_launch(void* const* d_in, const int* in_sizes, int n_in,
                              void* d_out, int out_size, void* d_ws, size_t ws_size,
                              hipStream_t stream) {
  (void)in_sizes; (void)n_in; (void)out_size; (void)ws_size;

  const float* x     = (const float*)d_in[0];
  const float* h0r   = (const float*)d_in[1];
  const float* h0i   = (const float*)d_in[2];
  const float* Ur_w  = (const float*)d_in[3];
  const float* Ur_b  = (const float*)d_in[4];
  const float* Ui_w  = (const float*)d_in[5];
  const float* Ui_b  = (const float*)d_in[6];
  const float* Wr_w  = (const float*)d_in[7];
  const float* Wr_b  = (const float*)d_in[8];
  const float* Wi_w  = (const float*)d_in[9];
  const float* Wi_b  = (const float*)d_in[10];
  const float* l1r_w = (const float*)d_in[11];
  const float* l1r_b = (const float*)d_in[12];
  const float* l1i_w = (const float*)d_in[13];
  const float* l1i_b = (const float*)d_in[14];
  const float* l2r_w = (const float*)d_in[15];
  const float* l2r_b = (const float*)d_in[16];
  const float* l2i_w = (const float*)d_in[17];
  const float* l2i_b = (const float*)d_in[18];
  const float* l3r_w = (const float*)d_in[19];
  const float* l3r_b = (const float*)d_in[20];
  const float* l3i_w = (const float*)d_in[21];
  const float* l3i_b = (const float*)d_in[22];
  const float* l5_w  = (const float*)d_in[23];
  const float* l5_b  = (const float*)d_in[24];

  // ---- workspace layout (bf16 buffers), 256B aligned ----
  char* ws = (char*)d_ws;
  size_t off = 0;
  auto alloc = [&](size_t bytes) -> size_t {
    size_t r = off;
    off = (off + bytes + 255) & ~(size_t)255;
    return r;
  };
  const size_t HH  = (size_t)HDIM * HDIM * 2;        // 768x768 bf16
  const size_t WH  = (size_t)W2DIM * HDIM * 2;       // 3072x768 bf16
  const size_t WW  = (size_t)W2DIM * W2DIM * 2;      // 3072x3072 bf16
  const size_t ACT = (size_t)MTOT * HDIM * 2;        // 16384x768 bf16
  const size_t HB  = (size_t)BDIM * HDIM * 2;        // 256x768 bf16
  const size_t MB  = (size_t)BDIM * W2DIM * 2;       // 256x3072 bf16

  bf16* wUr  = (bf16*)(ws + alloc(HH));
  bf16* wUi  = (bf16*)(ws + alloc(HH));
  bf16* wWr  = (bf16*)(ws + alloc(HH));
  bf16* wWi  = (bf16*)(ws + alloc(HH));
  bf16* wL1r = (bf16*)(ws + alloc(WH));
  bf16* wL1i = (bf16*)(ws + alloc(WH));
  bf16* wL2r = (bf16*)(ws + alloc(WW));
  bf16* wL2i = (bf16*)(ws + alloc(WW));
  bf16* wL3r = (bf16*)(ws + alloc(WW));
  bf16* wL3i = (bf16*)(ws + alloc(WW));
  bf16* xrB  = (bf16*)(ws + alloc(ACT));
  bf16* xiB  = (bf16*)(ws + alloc(ACT));
  bf16* crB  = (bf16*)(ws + alloc(ACT));
  bf16* ciB  = (bf16*)(ws + alloc(ACT));
  bf16* hR[2] = { (bf16*)(ws + alloc(HB)), (bf16*)(ws + alloc(HB)) };
  bf16* hI[2] = { (bf16*)(ws + alloc(HB)), (bf16*)(ws + alloc(HB)) };
  bf16* m1r  = (bf16*)(ws + alloc(MB));
  bf16* m1i  = (bf16*)(ws + alloc(MB));
  bf16* m2r  = (bf16*)(ws + alloc(MB));
  bf16* m2i  = (bf16*)(ws + alloc(MB));
  bf16* m3r  = (bf16*)(ws + alloc(MB));
  bf16* m3i  = (bf16*)(ws + alloc(MB));

  auto cvt = [&](const float* s, bf16* d, int n) {
    cvt_kernel<<<(n + 255) / 256, 256, 0, stream>>>(s, d, n);
  };

  // 1) one-time weight conversion to bf16 (stays resident in 192MB L2)
  cvt(Ur_w,  wUr,  HDIM * HDIM);
  cvt(Ui_w,  wUi,  HDIM * HDIM);
  cvt(Wr_w,  wWr,  HDIM * HDIM);
  cvt(Wi_w,  wWi,  HDIM * HDIM);
  cvt(l1r_w, wL1r, W2DIM * HDIM);
  cvt(l1i_w, wL1i, W2DIM * HDIM);
  cvt(l2r_w, wL2r, W2DIM * W2DIM);
  cvt(l2i_w, wL2i, W2DIM * W2DIM);
  cvt(l3r_w, wL3r, W2DIM * W2DIM);
  cvt(l3i_w, wL3i, W2DIM * W2DIM);
  cvt(h0r, hR[0], BDIM * HDIM);
  cvt(h0i, hI[0], BDIM * HDIM);

  // 2) split x into real/imag bf16 halves
  {
    int n = MTOT * HDIM;
    split_x_kernel<<<(n + 255) / 256, 256, 0, stream>>>(x, xrB, xiB);
  }

  // 3) fused U-projection: cr = xr*Ur^T - xi*Ui^T + (Ur_b - Ui_b),
  //                        ci = xi*Ur^T + xr*Ui^T + (Ur_b + Ui_b)
  {
    dim3 grid(HDIM / 64, MTOT / 64);  // 12 x 256
    cgemm_kernel<0><<<grid, 256, 0, stream>>>(
        xrB, xiB, wUr, wUi, Ur_b, Ui_b,
        nullptr, nullptr, 0, crB, ciB, MTOT, HDIM, HDIM);
  }

  // 4) recurrent scan: 64 dependent steps (latency chain)
  for (int t = 0; t < TDIM; ++t) {
    int cur = t & 1, nxt = cur ^ 1;
    dim3 grid(HDIM / 64, BDIM / 64);  // 12 x 4
    cgemm_kernel<1><<<grid, 256, 0, stream>>>(
        hR[cur], hI[cur], wWr, wWi, Wr_b, Wi_b,
        crB + (size_t)t * HDIM, ciB + (size_t)t * HDIM, TDIM * HDIM,
        hR[nxt], hI[nxt], BDIM, HDIM, HDIM);
  }
  // T=64 even -> final hidden state is in buffers [0]

  // 5) complex MLP layers (lrelu before combine)
  {
    dim3 grid(W2DIM / 64, BDIM / 64);  // 48 x 4
    cgemm_kernel<2><<<grid, 256, 0, stream>>>(
        hR[0], hI[0], wL1r, wL1i, l1r_b, l1i_b,
        nullptr, nullptr, 0, m1r, m1i, BDIM, W2DIM, HDIM);
    cgemm_kernel<2><<<grid, 256, 0, stream>>>(
        m1r, m1i, wL2r, wL2i, l2r_b, l2i_b,
        nullptr, nullptr, 0, m2r, m2i, BDIM, W2DIM, W2DIM);
    cgemm_kernel<2><<<grid, 256, 0, stream>>>(
        m2r, m2i, wL3r, wL3i, l3r_b, l3i_b,
        nullptr, nullptr, 0, m3r, m3i, BDIM, W2DIM, W2DIM);
  }

  // 6) final scalar head
  l5_kernel<<<BDIM, 256, 0, stream>>>(m3r, m3i, l5_w, l5_b, (float*)d_out);
}
